// NLM_2319282340669
// MI455X (gfx1250) — matrix operand order repaired and checked by reading, not compiled, not run on hardware
//
#include <hip/hip_runtime.h>

typedef __attribute__((ext_vector_type(16))) _Float16 v16h;
typedef __attribute__((ext_vector_type(8)))  float    v8f;
typedef __attribute__((ext_vector_type(4)))  float    f4;

#define T      32
#define NPIX   1024          // T*T pixels per tile
#define KREAL  72            // 3 channels * 24 non-center patch offsets
#define KDIM   96            // padded to 3 chunks of K=32
#define NBLK   64            // NPIX / 16

#define LOG2E  1.4426950408889634f

union AFrag { v16h v; f4 f[2]; };

__device__ __forceinline__ int iclamp(int v, int lo, int hi) {
    return v < lo ? lo : (v > hi ? hi : v);
}

// Load one m-block's three K-chunk A fragments (per ISA A layout:
// lane<16 holds K {0..7,16..23}; lane>=16 holds K {8..15,24..31}).
__device__ __forceinline__ void loadA(const _Float16* __restrict__ Yh,
                                      int mb, int lcol, int hi,
                                      AFrag& a0, AFrag& a1, AFrag& a2)
{
    const _Float16* ar = Yh + (mb * 16 + lcol) * KDIM + hi * 8;
    a0.f[0] = *(const f4*)(ar +  0); a0.f[1] = *(const f4*)(ar + 16);
    a1.f[0] = *(const f4*)(ar + 32); a1.f[1] = *(const f4*)(ar + 48);
    a2.f[0] = *(const f4*)(ar + 64); a2.f[1] = *(const f4*)(ar + 80);
}

// 6 WMMAs (two n-blocks sharing one m-block A set) + density accumulation.
__device__ __forceinline__ void processBlock(
    int mb, int hi, const f4* __restrict__ pix,
    const AFrag& a0, const AFrag& a1, const AFrag& a2,
    const AFrag& b00, const AFrag& b01, const AFrag& b02,
    const AFrag& b10, const AFrag& b11, const AFrag& b12,
    float sqn0, float sqn1,
    float& accW0, float& accX0, float& accY0, float& accZ0,
    float& accW1, float& accX1, float& accY1, float& accZ1)
{
    // G^T blocks: rows = m (in VGPRs), cols = n (across lanes)
    v8f g0 = {}, g1 = {};
    g0 = __builtin_amdgcn_wmma_f32_16x16x32_f16(false, a0.v, false, b00.v, (short)0, g0, false, false);
    g1 = __builtin_amdgcn_wmma_f32_16x16x32_f16(false, a0.v, false, b10.v, (short)0, g1, false, false);
    g0 = __builtin_amdgcn_wmma_f32_16x16x32_f16(false, a1.v, false, b01.v, (short)0, g0, false, false);
    g1 = __builtin_amdgcn_wmma_f32_16x16x32_f16(false, a1.v, false, b11.v, (short)0, g1, false, false);
    g0 = __builtin_amdgcn_wmma_f32_16x16x32_f16(false, a2.v, false, b02.v, (short)0, g0, false, false);
    g1 = __builtin_amdgcn_wmma_f32_16x16x32_f16(false, a2.v, false, b12.v, (short)0, g1, false, false);

    const int mbase = mb * 16 + hi * 8;          // C/D layout: lanes>=16 hold M = r+8
    #pragma unroll
    for (int r = 0; r < 8; ++r) {
        int m = mbase + r;
        f4 pm = pix[m];                          // {y0,y1,y2,||y_m||^2}
        // density = exp(-dist): 2*Hp^2*sigma^2 == 1.0 for these params.
        // Diagonal is NOT masked here; subtracted analytically after the loop.
        float d2a = pm.w + sqn0 - 2.0f * g0[r];
        float d2b = pm.w + sqn1 - 2.0f * g1[r];
        float da = __builtin_amdgcn_exp2f(-LOG2E * __builtin_amdgcn_sqrtf(fmaxf(d2a, 0.f)));
        float db = __builtin_amdgcn_exp2f(-LOG2E * __builtin_amdgcn_sqrtf(fmaxf(d2b, 0.f)));
        accW0 += da;  accX0 += da * pm.x;  accY0 += da * pm.y;  accZ0 += da * pm.z;
        accW1 += db;  accX1 += db * pm.x;  accY1 += db * pm.y;  accZ1 += db * pm.z;
    }
}

__global__ __launch_bounds__(256)
void nlm_tile_kernel(const float* __restrict__ img, float* __restrict__ out)
{
    extern __shared__ char smem[];
    _Float16* Yh   = (_Float16*)smem;                                // 1024*96*2 = 196608 B
    f4*       pix  = (f4*)(smem + NPIX * KDIM * 2);                  // 1024*16   =  16384 B
    float*    timg = (float*)(smem + NPIX * KDIM * 2 + NPIX * 16);   // 3*1024*4  =  12288 B

    const int tid    = threadIdx.x;
    const int lane   = tid & 31;
    const int wave   = tid >> 5;
    const int tileId = blockIdx.x;             // 0..99
    const int th     = tileId / 10;
    const int tw     = tileId % 10;

    // ---- Phase 0: cache the 3x32x32 tile in LDS (coalesced) ----
    for (int i = tid; i < 3 * NPIX; i += 256) {
        int ch = i >> 10, n = i & (NPIX - 1);
        int r = n >> 5, c = n & 31;
        timg[i] = img[(ch * 320 + th * 32 + r) * 320 + tw * 32 + c];
    }
    __syncthreads();

    // ---- Phase 1: build f16 feature matrix Y[n][k] and {y0,y1,y2,||y||^2} table ----
    for (int n = tid; n < NPIX; n += 256) {
        int r = n >> 5, c = n & 31;
        float sq = 0.f;
        _Float16* yrow = Yh + n * KDIM;
        #pragma unroll
        for (int ch = 0; ch < 3; ++ch) {
            const float* im = timg + ch * NPIX;
            #pragma unroll
            for (int o = 0; o < 24; ++o) {
                int idx = (o < 12) ? o : o + 1;        // skip patch center (5x5 -> idx 12)
                int di = idx / 5 - 2, dj = idx % 5 - 2;
                int rr = iclamp(r + di, 0, 31);
                int cc = iclamp(c + dj, 0, 31);        // replicate pad at tile edge
                float v = im[rr * 32 + cc];
                yrow[ch * 24 + o] = (_Float16)v;
                sq += v * v;
            }
        }
        #pragma unroll
        for (int k = KREAL; k < KDIM; ++k) yrow[k] = (_Float16)0.f;
        f4 p;
        p.x = timg[n]; p.y = timg[NPIX + n]; p.z = timg[2 * NPIX + n]; p.w = sq;
        pix[n] = p;
    }
    __syncthreads();

    // ---- Phase 2: fused Gram(WMMA) -> exp -> weighted accumulation ----
    const int hi   = lane >> 4;            // half-select per WMMA f16 operand layout
    const int lcol = lane & 15;
    const int nb0  = blockIdx.y * 16 + wave * 2;   // first of this wave's n-blocks
    const int n0   = nb0 * 16 + lcol;              // lane's output pixel, block 0
    const int n1   = n0 + 16;                      // lane's output pixel, block 1

    const f4 p0 = pix[n0];
    const f4 p1 = pix[n1];
    const float sqn0 = p0.w;
    const float sqn1 = p1.w;

    // B fragments (Y^T chunks) for both n-blocks. Lane<16: K {0..7,16..23}, lane>=16: K {8..15,24..31}.
    AFrag b00, b01, b02, b10, b11, b12;
    {
        const _Float16* yr0 = Yh + n0 * KDIM + hi * 8;
        b00.f[0] = *(const f4*)(yr0 +  0); b00.f[1] = *(const f4*)(yr0 + 16);
        b01.f[0] = *(const f4*)(yr0 + 32); b01.f[1] = *(const f4*)(yr0 + 48);
        b02.f[0] = *(const f4*)(yr0 + 64); b02.f[1] = *(const f4*)(yr0 + 80);
        const _Float16* yr1 = Yh + n1 * KDIM + hi * 8;
        b10.f[0] = *(const f4*)(yr1 +  0); b10.f[1] = *(const f4*)(yr1 + 16);
        b11.f[0] = *(const f4*)(yr1 + 32); b11.f[1] = *(const f4*)(yr1 + 48);
        b12.f[0] = *(const f4*)(yr1 + 64); b12.f[1] = *(const f4*)(yr1 + 80);
    }

    // Partial f16-precision Gram diagonal from the resident B fragments
    // (lane pair l / l+16 covers all K of column l; folded with the accumulators).
    float gd0 = 0.f, gd1 = 0.f;
    #pragma unroll
    for (int i = 0; i < 16; ++i) {
        float v0a = (float)b00.v[i], v0b = (float)b01.v[i], v0c = (float)b02.v[i];
        float v1a = (float)b10.v[i], v1b = (float)b11.v[i], v1c = (float)b12.v[i];
        gd0 += v0a * v0a + v0b * v0b + v0c * v0c;
        gd1 += v1a * v1a + v1b * v1b + v1c * v1c;
    }

    float accW0 = 0.f, accX0 = 0.f, accY0 = 0.f, accZ0 = 0.f;
    float accW1 = 0.f, accX1 = 0.f, accY1 = 0.f, accZ1 = 0.f;

    // m-loop unrolled x2 with ping-pong A-fragment sets: set B's loads are in
    // flight while set A's block executes, and vice versa -- no register copies.
    AFrag xa0, xa1, xa2, ya0, ya1, ya2;
    loadA(Yh, 0, lcol, hi, xa0, xa1, xa2);

    for (int mb = 0; mb < NBLK; mb += 2) {
        loadA(Yh, mb + 1, lcol, hi, ya0, ya1, ya2);
        processBlock(mb, hi, pix, xa0, xa1, xa2,
                     b00, b01, b02, b10, b11, b12, sqn0, sqn1,
                     accW0, accX0, accY0, accZ0, accW1, accX1, accY1, accZ1);

        loadA(Yh, (mb + 2) & (NBLK - 1), lcol, hi, xa0, xa1, xa2);
        processBlock(mb + 1, hi, pix, ya0, ya1, ya2,
                     b00, b01, b02, b10, b11, b12, sqn0, sqn1,
                     accW0, accX0, accY0, accZ0, accW1, accX1, accY1, accZ1);
    }

    // lanes l and l+16 hold the two M-halves of the same column n -> fold
    accW0 += __shfl_xor(accW0, 16, 32);
    accX0 += __shfl_xor(accX0, 16, 32);
    accY0 += __shfl_xor(accY0, 16, 32);
    accZ0 += __shfl_xor(accZ0, 16, 32);
    accW1 += __shfl_xor(accW1, 16, 32);
    accX1 += __shfl_xor(accX1, 16, 32);
    accY1 += __shfl_xor(accY1, 16, 32);
    accZ1 += __shfl_xor(accZ1, 16, 32);
    gd0   += __shfl_xor(gd0,   16, 32);    // full f16 Gram diagonal of column n
    gd1   += __shfl_xor(gd1,   16, 32);

    if (lane < 16) {
        // Remove the diagonal's contribution: d_diag = exp(-sqrt(max(2*sqn-2*gd,0)))
        {
            float dd  = __builtin_amdgcn_exp2f(
                          -LOG2E * __builtin_amdgcn_sqrtf(fmaxf(2.f * sqn0 - 2.f * gd0, 0.f)));
            float w   = accW0 - dd;
            float inv = __builtin_amdgcn_rcpf(w);
            int r = n0 >> 5, c = n0 & 31;
            int o = (th * 32 + r) * 320 + tw * 32 + c;
            out[o]              = (accX0 - dd * p0.x) * inv;
            out[102400 + o]     = (accY0 - dd * p0.y) * inv;
            out[2 * 102400 + o] = (accZ0 - dd * p0.z) * inv;
        }
        {
            float dd  = __builtin_amdgcn_exp2f(
                          -LOG2E * __builtin_amdgcn_sqrtf(fmaxf(2.f * sqn1 - 2.f * gd1, 0.f)));
            float w   = accW1 - dd;
            float inv = __builtin_amdgcn_rcpf(w);
            int r = n1 >> 5, c = n1 & 31;
            int o = (th * 32 + r) * 320 + tw * 32 + c;
            out[o]              = (accX1 - dd * p1.x) * inv;
            out[102400 + o]     = (accY1 - dd * p1.y) * inv;
            out[2 * 102400 + o] = (accZ1 - dd * p1.z) * inv;
        }
    }
}

extern "C" void kernel_launch(void* const* d_in, const int* in_sizes, int n_in,
                              void* d_out, int out_size, void* d_ws, size_t ws_size,
                              hipStream_t stream) {
    (void)in_sizes; (void)n_in; (void)out_size; (void)d_ws; (void)ws_size;
    const float* img = (const float*)d_in[0];
    float* out = (float*)d_out;

    dim3 grid(100, 4, 1);       // 100 tiles x 4 row-quarters = 400 workgroups
    dim3 block(256, 1, 1);      // 8 wave32s
    size_t shmem = (size_t)NPIX * KDIM * 2   // f16 feature matrix
                 + (size_t)NPIX * 16         // pixel/sq table
                 + (size_t)3 * NPIX * 4;     // cached tile
    nlm_tile_kernel<<<grid, block, shmem, stream>>>(img, out);
}
